// TesseraAttention_26414048870806
// MI455X (gfx1250) — matrix operand order repaired and checked by reading, not compiled, not run on hardware
//
#include <hip/hip_runtime.h>

// ---------------- constants ----------------
constexpr int Bc   = 2;
constexpr int Tc   = 2048;
constexpr int HID  = 2048;
constexpr int Hc   = 16;
constexpr int KVH  = 4;
constexpr int Dc   = 128;
constexpr int BT   = Bc * Tc;          // 4096
constexpr int NQKV = Hc * Dc + 2 * KVH * Dc; // 3072
constexpr int NKV  = KVH * Dc;         // 512

// ---------------- CDNA5 vector types ----------------
typedef __bf16 v16bf __attribute__((ext_vector_type(16)));
typedef __bf16 v8bf  __attribute__((ext_vector_type(8)));
typedef float  v8f   __attribute__((ext_vector_type(8)));

__device__ __forceinline__ unsigned short f2bf(float f) {
    unsigned int u = __float_as_uint(f);
    unsigned int r = u + 0x7fffu + ((u >> 16) & 1u);   // round-to-nearest-even
    return (unsigned short)(r >> 16);
}

__device__ __forceinline__ v16bf cat8(v8bf lo, v8bf hi) {
    v16bf r;
#pragma unroll
    for (int i = 0; i < 8; ++i) { r[i] = lo[i]; r[i + 8] = hi[i]; }
    return r;
}

// ---------------- format conversion ----------------
__global__ void conv_bf16_kernel(const float* __restrict__ in,
                                 unsigned short* __restrict__ out, int n) {
    int i = blockIdx.x * blockDim.x + threadIdx.x;
    for (; i < n; i += gridDim.x * blockDim.x) out[i] = f2bf(in[i]);
}

// in: K x N (row-major fp32)  ->  out: N x K (row-major bf16) == B^T
__global__ void transpose_bf16_kernel(const float* __restrict__ in,
                                      unsigned short* __restrict__ out,
                                      int K, int N) {
    int i = blockIdx.x * blockDim.x + threadIdx.x;
    int total = K * N;
    for (; i < total; i += gridDim.x * blockDim.x) {
        int k = i / N, n = i % N;
        out[(size_t)n * K + k] = f2bf(in[i]);
    }
}

// ---------------- generic bf16 GEMM: C(f32, MxN) = A(bf16, MxK) * Bt(bf16, NxK)^T
__global__ __launch_bounds__(128)
void gemm_bf16_kernel(const unsigned short* __restrict__ A,
                      const unsigned short* __restrict__ Bt,
                      float* __restrict__ C, int M, int N, int K) {
    const int lane = threadIdx.x & 31;
    const int wave = threadIdx.x >> 5;
    const int lm   = lane & 15;
    const int hi   = lane >> 4;           // 0 or 1 (half-wave)
    const int m0   = blockIdx.y * 64 + wave * 16;
    const int n0   = blockIdx.x * 64;

    const unsigned short* arow = A + (size_t)(m0 + lm) * K;
    const int klo   = hi ? 8 : 0;
    const int klo16 = hi ? 16 : 0;

    v8f acc[4] = {};
    for (int k0 = 0; k0 < K; k0 += 32) {
        v16bf a = *reinterpret_cast<const v16bf*>(arow + k0 + klo16);
#pragma unroll
        for (int j = 0; j < 4; ++j) {
            const unsigned short* bp = Bt + (size_t)(n0 + j * 16 + lm) * K + k0 + klo16;
            v16bf bv = *reinterpret_cast<const v16bf*>(bp);
            acc[j] = __builtin_amdgcn_wmma_f32_16x16x32_bf16(
                false, a, false, bv, (short)0, acc[j], false, false);
        }
    }
#pragma unroll
    for (int j = 0; j < 4; ++j)
#pragma unroll
        for (int r = 0; r < 8; ++r)
            C[(size_t)(m0 + r + hi * 8) * N + n0 + j * 16 + lm] = acc[j][r];
}

// ---------------- RoPE + pack ----------------
// qkv fp32 row layout: [q: h*128+d | k: 2048+kvh*128+d | v: 2560+kvh*128+d]
__global__ void rope_q_kernel(const float* __restrict__ qkv,
                              const float* __restrict__ cosT,
                              const float* __restrict__ sinT,
                              unsigned short* __restrict__ qb) {
    int i = blockIdx.x * blockDim.x + threadIdx.x;        // B*T*H*64 = 2^22
    int d = i & 63, h = (i >> 6) & 15, t = (i >> 10) & 2047, b = i >> 21;
    const float* base = qkv + (size_t)(b * Tc + t) * NQKV + h * Dc;
    float x1 = base[d], x2 = base[64 + d];
    float c1 = cosT[t * Dc + d],      s1 = sinT[t * Dc + d];
    float c2 = cosT[t * Dc + 64 + d], s2 = sinT[t * Dc + 64 + d];
    const float sc = 0.08838834764831845f;                // 1/sqrt(128)
    unsigned short* q = qb + (size_t)((b * Tc + t) * Hc + h) * Dc;
    q[d]      = f2bf((x1 * c1 - x2 * s1) * sc);
    q[64 + d] = f2bf((x2 * c2 + x1 * s2) * sc);
}

__global__ void rope_k_kernel(const float* __restrict__ qkv,
                              const float* __restrict__ cosT,
                              const float* __restrict__ sinT,
                              unsigned short* __restrict__ kb) {
    int i = blockIdx.x * blockDim.x + threadIdx.x;        // B*T*KVH*64 = 2^20
    int d = i & 63, kvh = (i >> 6) & 3, t = (i >> 8) & 2047, b = i >> 19;
    const float* base = qkv + (size_t)(b * Tc + t) * NQKV + Hc * Dc + kvh * Dc;
    float x1 = base[d], x2 = base[64 + d];
    float c1 = cosT[t * Dc + d],      s1 = sinT[t * Dc + d];
    float c2 = cosT[t * Dc + 64 + d], s2 = sinT[t * Dc + 64 + d];
    unsigned short* k = kb + (size_t)((b * Tc + t) * KVH + kvh) * Dc;
    k[d]      = f2bf(x1 * c1 - x2 * s1);
    k[64 + d] = f2bf(x2 * c2 + x1 * s2);
}

// v -> bf16, transposed to (b, kvh, d, t) so PV B-operands are contiguous
__global__ void conv_v_kernel(const float* __restrict__ qkv,
                              unsigned short* __restrict__ vt) {
    int i = blockIdx.x * blockDim.x + threadIdx.x;        // B*T*KVH*128 = 2^21
    int d = i & 127, kvh = (i >> 7) & 3, t = (i >> 9) & 2047, b = i >> 20;
    float val = qkv[(size_t)(b * Tc + t) * NQKV + Hc * Dc + NKV + kvh * Dc + d];
    vt[((size_t)(b * KVH + kvh) * Dc + d) * Tc + t] = f2bf(val);
}

// ---------------- flash attention (causal, GQA) ----------------
__global__ __launch_bounds__(128)
void flash_attn_kernel(const unsigned short* __restrict__ qb,
                       const unsigned short* __restrict__ kb,
                       const unsigned short* __restrict__ vt,
                       unsigned short* __restrict__ attn) {
    __shared__ unsigned short pbuf[4][16 * 32];           // per-wave P tile (bf16)
    const int lane = threadIdx.x & 31;
    const int wave = threadIdx.x >> 5;
    const int lm   = lane & 15;
    const int hi   = lane >> 4;
    const int b    = blockIdx.z;
    const int h    = blockIdx.y;
    const int kvh  = h >> 2;                              // H/KVH = 4
    const int q0   = (blockIdx.x * 4 + wave) * 16;
    const float neg_inf = -__builtin_inff();

    // Q A-operands (16x128 -> 4 chunks of 16x32), pre-scaled by 1/sqrt(D)
    v16bf aq[4];
    {
        const unsigned short* qrow = qb + (size_t)((b * Tc + q0 + lm) * Hc + h) * Dc;
        const int klo = hi ? 8 : 0;
#pragma unroll
        for (int cc = 0; cc < 4; ++cc) {
            aq[cc] = *reinterpret_cast<const v16bf*>(qrow + cc * 32 + 2 * klo);
        }
    }

    v8f acc[8] = {};                                      // O tile 16x128
    float m[8], l[8];
#pragma unroll
    for (int r = 0; r < 8; ++r) { m[r] = neg_inf; l[r] = 0.f; }

    const unsigned short* kbp = kb + ((size_t)(b * Tc) * KVH + kvh) * Dc;
    const unsigned short* vbp = vt + (size_t)(b * KVH + kvh) * Dc * Tc;
    const int klo   = hi ? 8 : 0;
    const int klo16 = hi ? 16 : 0;
    const int ntiles = (q0 + 47) >> 5;                    // cover keys 0..q0+15

    for (int kt = 0; kt < ntiles; ++kt) {
        const int k0 = kt * 32;
        // S = Q * K^T  (two 16x16 key tiles)
        v8f s0 = {}, s1 = {};
#pragma unroll
        for (int cc = 0; cc < 4; ++cc) {
            const int d = cc * 32 + klo16;
            v16bf bk0 = *reinterpret_cast<const v16bf*>(kbp + (size_t)(k0 + lm) * NKV + d);
            v16bf bk1 = *reinterpret_cast<const v16bf*>(kbp + (size_t)(k0 + 16 + lm) * NKV + d);
            s0 = __builtin_amdgcn_wmma_f32_16x16x32_bf16(false, aq[cc], false, bk0, (short)0, s0, false, false);
            s1 = __builtin_amdgcn_wmma_f32_16x16x32_bf16(false, aq[cc], false, bk1, (short)0, s1, false, false);
        }
        // causal mask on boundary tiles
        if (k0 + 31 > q0) {
#pragma unroll
            for (int r = 0; r < 8; ++r) {
                int row = q0 + r + hi * 8;
                s0[r] = (k0 + lm > row)      ? neg_inf : s0[r];
                s1[r] = (k0 + 16 + lm > row) ? neg_inf : s1[r];
            }
        }
        // online softmax (row reductions stay inside 16-lane halves)
#pragma unroll
        for (int r = 0; r < 8; ++r) {
            float mt = fmaxf(s0[r], s1[r]);
            mt = fmaxf(mt, __shfl_xor(mt, 1));
            mt = fmaxf(mt, __shfl_xor(mt, 2));
            mt = fmaxf(mt, __shfl_xor(mt, 4));
            mt = fmaxf(mt, __shfl_xor(mt, 8));
            float mn   = fmaxf(m[r], mt);
            float corr = __expf(m[r] - mn);
            m[r] = mn;
            float p0 = __expf(s0[r] - mn);
            float p1 = __expf(s1[r] - mn);
            s0[r] = p0; s1[r] = p1;
            float rs = p0 + p1;
            rs += __shfl_xor(rs, 1);
            rs += __shfl_xor(rs, 2);
            rs += __shfl_xor(rs, 4);
            rs += __shfl_xor(rs, 8);
            l[r] = l[r] * corr + rs;
#pragma unroll
            for (int n = 0; n < 8; ++n) acc[n][r] *= corr;
        }
        // re-layout P (C-layout -> A-layout) through per-wave LDS
        unsigned short* pb = pbuf[wave];
#pragma unroll
        for (int r = 0; r < 8; ++r) {
            int row = r + hi * 8;
            pb[row * 32 + lm]      = f2bf(s0[r]);
            pb[row * 32 + 16 + lm] = f2bf(s1[r]);
        }
        asm volatile("s_wait_dscnt 0x0" ::: "memory");
        v16bf ap;
        {
            ap = *reinterpret_cast<const v16bf*>(pb + lm * 32 + klo16);
        }
        // O += P * V  (v stored (b,kvh,d,t): B-operand is one contiguous load)
#pragma unroll
        for (int n = 0; n < 8; ++n) {
            const unsigned short* vp = vbp + (size_t)(n * 16 + lm) * Tc + k0 + klo16;
            v16bf bv = *reinterpret_cast<const v16bf*>(vp);
            acc[n] = __builtin_amdgcn_wmma_f32_16x16x32_bf16(
                false, ap, false, bv, (short)0, acc[n], false, false);
        }
    }

    // normalize and emit bf16 (b, t, h*D)
#pragma unroll
    for (int r = 0; r < 8; ++r) {
        float inv = 1.f / l[r];
        int row = q0 + r + hi * 8;
        unsigned short* orow = attn + (size_t)(b * Tc + row) * (Hc * Dc) + h * Dc;
#pragma unroll
        for (int n = 0; n < 8; ++n) orow[n * 16 + lm] = f2bf(acc[n][r] * inv);
    }
}

// ---------------- launch ----------------
extern "C" void kernel_launch(void* const* d_in, const int* in_sizes, int n_in,
                              void* d_out, int out_size, void* d_ws, size_t ws_size,
                              hipStream_t stream) {
    const float* x  = (const float*)d_in[0];
    const float* rc = (const float*)d_in[1];
    const float* rs = (const float*)d_in[2];
    const float* Wq = (const float*)d_in[3];
    const float* Wk = (const float*)d_in[4];
    const float* Wv = (const float*)d_in[5];
    const float* Wo = (const float*)d_in[6];
    float* out = (float*)d_out;

    char* ws = (char*)d_ws;
    size_t off = 0;
    auto alloc = [&](size_t bytes) -> char* {
        char* p = ws + off;
        off += (bytes + 255) & ~(size_t)255;
        return p;
    };
    unsigned short* xb   = (unsigned short*)alloc((size_t)BT * HID * 2);       // x bf16
    unsigned short* Wt   = (unsigned short*)alloc((size_t)NQKV * HID * 2);     // [Wq;Wk;Wv]^T
    unsigned short* Wot  = (unsigned short*)alloc((size_t)HID * (Hc * Dc) * 2);
    float*          qkv  = (float*)alloc((size_t)BT * NQKV * 4);
    unsigned short* qbuf = (unsigned short*)alloc((size_t)BT * Hc * Dc * 2);
    unsigned short* kbuf = (unsigned short*)alloc((size_t)BT * KVH * Dc * 2);
    unsigned short* vtb  = (unsigned short*)alloc((size_t)BT * KVH * Dc * 2);
    unsigned short* atb  = (unsigned short*)alloc((size_t)BT * Hc * Dc * 2);

    const int TPB = 256;
    // 1) convert / transpose to bf16
    conv_bf16_kernel<<<(BT * HID) / TPB, TPB, 0, stream>>>(x, xb, BT * HID);
    transpose_bf16_kernel<<<(HID * Hc * Dc) / TPB, TPB, 0, stream>>>(Wq, Wt, HID, Hc * Dc);
    transpose_bf16_kernel<<<(HID * NKV) / TPB, TPB, 0, stream>>>(Wk, Wt + (size_t)(Hc * Dc) * HID, HID, NKV);
    transpose_bf16_kernel<<<(HID * NKV) / TPB, TPB, 0, stream>>>(Wv, Wt + (size_t)(Hc * Dc + NKV) * HID, HID, NKV);
    transpose_bf16_kernel<<<(Hc * Dc * HID) / TPB, TPB, 0, stream>>>(Wo, Wot, Hc * Dc, HID);

    // 2) fused QKV projection (WMMA)
    gemm_bf16_kernel<<<dim3(NQKV / 64, BT / 64), 128, 0, stream>>>(xb, Wt, qkv, BT, NQKV, HID);

    // 3) RoPE + pack
    rope_q_kernel<<<(BT * Hc * 64) / TPB, TPB, 0, stream>>>(qkv, rc, rs, qbuf);
    rope_k_kernel<<<(BT * KVH * 64) / TPB, TPB, 0, stream>>>(qkv, rc, rs, kbuf);
    conv_v_kernel<<<(BT * KVH * Dc) / TPB, TPB, 0, stream>>>(qkv, vtb);

    // 4) causal flash attention (WMMA)
    flash_attn_kernel<<<dim3(Tc / 64, Hc, Bc), 128, 0, stream>>>(qbuf, kbuf, vtb, atb);

    // 5) output projection (WMMA)
    gemm_bf16_kernel<<<dim3(HID / 64, BT / 64), 128, 0, stream>>>(atb, Wot, out, BT, HID, Hc * Dc);
}
